// GATStack_72378788872287
// MI455X (gfx1250) — matrix operand order repaired and checked by reading, not compiled, not run on hardware
//
#include <hip/hip_runtime.h>
#include <stdint.h>

// ---------------------------------------------------------------------------
// GAT stack for MI455X (gfx1250).
//  * GEMMs: v_wmma_f32_16x16x32_bf16 (fp32 accum); weights pre-transposed to
//    bf16, activations converted fp32->bf16 in-register.
//  * Aggregation: CSR-by-dst built once per edge set (int atomics + LDS scan),
//    then gather-based softmax/weighted-sum. This replaces ~330M fp32 atomic
//    RMWs (slow L2 atomic path) with coalesced float4 gathers; node features
//    (<=82MB) are L2-resident (192MB), so the stage runs at L2 load bandwidth.
// ---------------------------------------------------------------------------

#define NNODES 20000
#define NEDGES 320000
#define NHEADS 4
#define SLOPE  0.2f

typedef __attribute__((ext_vector_type(16))) __bf16 v16bf;
typedef __attribute__((ext_vector_type(8)))  float  v8f;

// ---------------- utility kernels ------------------------------------------
__global__ void fill_i32_kernel(int* __restrict__ p, int v, long n) {
  long i = (long)blockIdx.x * 256 + threadIdx.x;
  if (i < n) p[i] = v;
}

// W[K][ncols] fp32 (row-major)  ->  Wt[ncolsPad][K] bf16 (zero-padded cols)
__global__ void wtrans_kernel(const float* __restrict__ W, __bf16* __restrict__ Wt,
                              int K, int ncols, int ncolsPad) {
  long i = (long)blockIdx.x * 256 + threadIdx.x;
  long tot = (long)ncolsPad * K;
  if (i >= tot) return;
  int n = (int)(i / K), k = (int)(i % K);
  float v = (n < ncols) ? W[(size_t)k * ncols + n] : 0.f;
  Wt[(size_t)n * K + k] = (__bf16)v;
}

// ---------------- WMMA GEMM: C[M,ncols] = A[M,K] * W  (+bias) --------------
// Block = 256 threads = 8 waves laid out 2(M)x4(N); block tile 32x64.
__global__ __launch_bounds__(256)
void gemm_bf16_wmma(const float* __restrict__ A, const __bf16* __restrict__ Wt,
                    float* __restrict__ C, const float* __restrict__ bias,
                    int K, int ncols, int ldc) {
  const int wave  = threadIdx.x >> 5;
  const int lane  = threadIdx.x & 31;
  const int l15   = lane & 15;
  const int lhalf = lane >> 4;               // 0 | 1
  const int row0  = blockIdx.x * 32 + (wave >> 2) * 16;
  const int col0  = blockIdx.y * 64 + (wave & 3) * 16;

  const float*  arow = A  + (size_t)(row0 + l15) * K;
  const __bf16* brow = Wt + (size_t)(col0 + l15) * K + 16 * lhalf;

  v8f acc = {0.f, 0.f, 0.f, 0.f, 0.f, 0.f, 0.f, 0.f};

  for (int k0 = 0; k0 < K; k0 += 32) {
    // A frag (16x32 bf16): lane half selects K blocks {0-15} / {16-31}
    const int kA = k0 + 16 * lhalf;
    const float4* ap0 = (const float4*)(arow + kA);        // K = kA .. kA+7
    const float4* ap1 = (const float4*)(arow + kA + 8);    // K = kA+8 .. kA+15
    float4 a0 = ap0[0], a1 = ap0[1], a2 = ap1[0], a3 = ap1[1];
    v16bf af;
    af[0]=(__bf16)a0.x; af[1]=(__bf16)a0.y; af[2]=(__bf16)a0.z; af[3]=(__bf16)a0.w;
    af[4]=(__bf16)a1.x; af[5]=(__bf16)a1.y; af[6]=(__bf16)a1.z; af[7]=(__bf16)a1.w;
    af[8]=(__bf16)a2.x; af[9]=(__bf16)a2.y; af[10]=(__bf16)a2.z; af[11]=(__bf16)a2.w;
    af[12]=(__bf16)a3.x; af[13]=(__bf16)a3.y; af[14]=(__bf16)a3.z; af[15]=(__bf16)a3.w;
    // B frag (32x16 bf16): lanes 0-15 K=k0..k0+15, lanes 16-31 K=k0+16..k0+31
    v16bf bf = *(const v16bf*)(brow + k0);
    acc = __builtin_amdgcn_wmma_f32_16x16x32_bf16(false, af, false, bf,
                                                  (short)0, acc, false, false);
  }

  const int n = col0 + l15;
  if (n < ncols) {
    const float b = bias ? bias[n] : 0.f;
    const int mbase = row0 + 8 * lhalf;
#pragma unroll
    for (int r = 0; r < 8; ++r)
      C[(size_t)(mbase + r) * ldc + n] = acc[r] + b;
  }
}

// ---------------- CSR-by-destination build ---------------------------------
__global__ void count_kernel(const int* __restrict__ dst, int* __restrict__ cnt) {
  int e = blockIdx.x * 256 + threadIdx.x;
  if (e < NEDGES) atomicAdd(&cnt[dst[e]], 1);
}

// single-block exclusive prefix scan over cnt[0..n) -> rowstart / cursor
__global__ __launch_bounds__(1024)
void scan_kernel(const int* __restrict__ cnt, int* __restrict__ rowstart,
                 int* __restrict__ cursor, int n, int total) {
  __shared__ int sh[1024];
  const int t = threadIdx.x;
  const int chunk = (n + 1023) / 1024;
  const int base = t * chunk;
  int lsum = 0;
  for (int i = 0; i < chunk; ++i) {
    int idx = base + i;
    if (idx < n) lsum += cnt[idx];
  }
  sh[t] = lsum;
  __syncthreads();
  for (int off = 1; off < 1024; off <<= 1) {   // Hillis-Steele inclusive scan
    int v = (t >= off) ? sh[t - off] : 0;
    __syncthreads();
    sh[t] += v;
    __syncthreads();
  }
  int run = sh[t] - lsum;                      // exclusive prefix of this chunk
  for (int i = 0; i < chunk; ++i) {
    int idx = base + i;
    if (idx < n) {
      rowstart[idx] = run;
      cursor[idx]   = run;
      run += cnt[idx];
    }
  }
  if (t == 1023) rowstart[n] = total;
}

__global__ void fillcsr_kernel(const int* __restrict__ dst, int* __restrict__ cursor,
                               int* __restrict__ eidx) {
  int e = blockIdx.x * 256 + threadIdx.x;
  if (e < NEDGES) {
    int p = atomicAdd(&cursor[dst[e]], 1);
    eidx[p] = e;
  }
}

// ---------------- attention projections el/er ------------------------------
__global__ void attn_proj_kernel(const float* __restrict__ f,
                                 const float* __restrict__ al,
                                 const float* __restrict__ ar,
                                 float* __restrict__ el, float* __restrict__ er,
                                 int D) {
  int i = blockIdx.x * 256 + threadIdx.x;          // over N*H
  if (i >= NNODES * NHEADS) return;
  int n = i >> 2, h = i & 3;
  const float* fp  = f  + (size_t)n * (NHEADS * D) + (size_t)h * D;
  const float* alp = al + (size_t)h * D;
  const float* arp = ar + (size_t)h * D;
  float sl = 0.f, sr = 0.f;
  for (int d = 0; d < D; d += 4) {
    float4 fv = *(const float4*)(fp + d);
    float4 av = *(const float4*)(alp + d);
    float4 bv = *(const float4*)(arp + d);
    sl += fv.x*av.x + fv.y*av.y + fv.z*av.z + fv.w*av.w;
    sr += fv.x*bv.x + fv.y*bv.y + fv.z*bv.z + fv.w*bv.w;
  }
  el[i] = sl; er[i] = sr;
}

// ---------------- per-edge score (pure, no atomics) ------------------------
__global__ void edge_score_kernel(const int* __restrict__ src, const int* __restrict__ dst,
                                  const float* __restrict__ el, const float* __restrict__ er,
                                  float* __restrict__ ebuf) {
  int i = blockIdx.x * 256 + threadIdx.x;          // over E*H
  if (i >= NEDGES * NHEADS) return;
  int e = i >> 2, h = i & 3;
  float x = el[src[e] * NHEADS + h] + er[dst[e] * NHEADS + h];
  ebuf[i] = (x > 0.f) ? x : SLOPE * x;             // leaky_relu
}

// ---------------- per-(node,head) softmax over incoming edges --------------
__global__ void node_softmax_kernel(const int* __restrict__ rowstart,
                                    const int* __restrict__ eidx,
                                    const float* __restrict__ ebuf,
                                    float* __restrict__ abuf, float* __restrict__ den) {
  int i = blockIdx.x * 256 + threadIdx.x;          // over N*H
  if (i >= NNODES * NHEADS) return;
  int n = i >> 2, h = i & 3;
  int j0 = rowstart[n], j1 = rowstart[n + 1];
  float m = -3.402823e38f;
  for (int j = j0; j < j1; ++j)
    m = fmaxf(m, ebuf[eidx[j] * NHEADS + h]);
  if (j1 == j0) m = 0.f;                           // isolated dst -> 0
  float s = 0.f;
  for (int j = j0; j < j1; ++j) {
    int e = eidx[j];
    float aa = __expf(ebuf[e * NHEADS + h] - m);
    abuf[e * NHEADS + h] = aa;                     // each edge has one dst: no race
    s += aa;
  }
  den[i] = s;
}

// ---------------- gather aggregation + normalize/res/elu/accumulate --------
__global__ void node_aggr_kernel(const int* __restrict__ rowstart, const int* __restrict__ eidx,
                                 const int* __restrict__ src,
                                 const float* __restrict__ abuf, const float* __restrict__ f,
                                 const float* __restrict__ den, const float* __restrict__ res,
                                 float* __restrict__ out,
                                 int D, int HD, float scale, int accumulate, int act) {
  const int qpn = HD >> 2;                         // float4 chunks per node
  long i = (long)blockIdx.x * 256 + threadIdx.x;   // over N * qpn
  if (i >= (long)NNODES * qpn) return;
  int n  = (int)(i / qpn);
  int hd = (int)(i % qpn) * 4;
  int h  = hd / D;
  int j0 = rowstart[n], j1 = rowstart[n + 1];
  float4 acc = {0.f, 0.f, 0.f, 0.f};
  for (int j = j0; j < j1; ++j) {
    int e = eidx[j];
    float aa = abuf[e * NHEADS + h];
    float4 fv = *(const float4*)(f + (size_t)src[e] * HD + hd);
    acc.x += aa * fv.x; acc.y += aa * fv.y;
    acc.z += aa * fv.z; acc.w += aa * fv.w;
  }
  float dv = fmaxf(den[n * NHEADS + h], 1e-9f);
  float4 v = {acc.x / dv, acc.y / dv, acc.z / dv, acc.w / dv};
  size_t o = (size_t)n * HD + hd;
  if (res) {
    float4 rv = *(const float4*)(res + o);
    v.x += rv.x; v.y += rv.y; v.z += rv.z; v.w += rv.w;
  }
  if (act) {                                       // elu
    v.x = (v.x > 0.f) ? v.x : (__expf(v.x) - 1.f);
    v.y = (v.y > 0.f) ? v.y : (__expf(v.y) - 1.f);
    v.z = (v.z > 0.f) ? v.z : (__expf(v.z) - 1.f);
    v.w = (v.w > 0.f) ? v.w : (__expf(v.w) - 1.f);
  }
  v.x *= scale; v.y *= scale; v.z *= scale; v.w *= scale;
  float* op = out + o;
  if (accumulate) {
    float4 ov = *(const float4*)op;
    v.x += ov.x; v.y += ov.y; v.z += ov.z; v.w += ov.w;
  }
  *(float4*)op = v;
}

// ---------------------------------------------------------------------------
// host-side orchestration
// ---------------------------------------------------------------------------
namespace {

struct Scratch {
  float *f, *res, *bufA, *bufB, *el, *er, *den, *ebuf, *abuf;
  int *cnt, *rs_g, *eidx_g, *rs_s, *eidx_s;
  __bf16* wt;
};

inline size_t alignup(size_t x) { return (x + 255) & ~size_t(255); }
inline dim3 g1(long n) { return dim3((unsigned)((n + 255) / 256)); }

void build_csr(const int* dst, int* cnt, int* rowstart, int* eidx, hipStream_t st) {
  fill_i32_kernel<<<g1(NNODES), 256, 0, st>>>(cnt, 0, NNODES);
  count_kernel<<<g1(NEDGES), 256, 0, st>>>(dst, cnt);
  scan_kernel<<<1, 1024, 0, st>>>(cnt, rowstart, cnt /*cursor reuse*/, NNODES, NEDGES);
  // NOTE: scan writes cursor==rowstart into cnt, so cnt now acts as cursor.
  fillcsr_kernel<<<g1(NEDGES), 256, 0, st>>>(dst, cnt, eidx);
}

void run_branch(const float* hin, int K, int D,
                const float* W, const float* al, const float* ar,
                const float* resW,
                const int* src, const int* dst,
                const int* rowstart, const int* eidx,
                float* out, float scale, int accumulate, int act,
                const Scratch& ws, hipStream_t st) {
  const int HD = NHEADS * D;                    // 256 or 1024 (mult of 64)
  wtrans_kernel<<<g1((long)HD * K), 256, 0, st>>>(W, ws.wt, K, HD, HD);
  gemm_bf16_wmma<<<dim3(NNODES / 32, HD / 64), 256, 0, st>>>(hin, ws.wt, ws.f, nullptr, K, HD, HD);
  if (resW) {
    wtrans_kernel<<<g1((long)HD * K), 256, 0, st>>>(resW, ws.wt, K, HD, HD);
    gemm_bf16_wmma<<<dim3(NNODES / 32, HD / 64), 256, 0, st>>>(hin, ws.wt, ws.res, nullptr, K, HD, HD);
  }
  attn_proj_kernel<<<g1((long)NNODES * NHEADS), 256, 0, st>>>(ws.f, al, ar, ws.el, ws.er, D);
  edge_score_kernel<<<g1((long)NEDGES * NHEADS), 256, 0, st>>>(src, dst, ws.el, ws.er, ws.ebuf);
  node_softmax_kernel<<<g1((long)NNODES * NHEADS), 256, 0, st>>>(rowstart, eidx, ws.ebuf,
                                                                 ws.abuf, ws.den);
  node_aggr_kernel<<<g1((long)NNODES * (HD / 4)), 256, 0, st>>>(rowstart, eidx, src, ws.abuf,
                                                                ws.f, ws.den,
                                                                resW ? ws.res : nullptr,
                                                                out, D, HD, scale, accumulate, act);
}

} // namespace

extern "C" void kernel_launch(void* const* d_in, const int* in_sizes, int n_in,
                              void* d_out, int out_size, void* d_ws, size_t ws_size,
                              hipStream_t stream) {
  (void)in_sizes; (void)n_in; (void)out_size; (void)ws_size;
  const float* x     = (const float*)d_in[0];
  const int*  src_g  = (const int*)d_in[1];
  const int*  dst_g  = (const int*)d_in[2];
  const int*  src_s  = (const int*)d_in[3];
  const int*  dst_s  = (const int*)d_in[4];
  const float *Wg0=(const float*)d_in[5],  *alg0=(const float*)d_in[6],  *arg0=(const float*)d_in[7];
  const float *Wg1=(const float*)d_in[8],  *alg1=(const float*)d_in[9],  *arg1=(const float*)d_in[10];
  const float *Wg2=(const float*)d_in[11], *alg2=(const float*)d_in[12], *arg2=(const float*)d_in[13];
  const float *Ws0=(const float*)d_in[14], *als0=(const float*)d_in[15], *ars0=(const float*)d_in[16];
  const float *Ws1=(const float*)d_in[17], *als1=(const float*)d_in[18], *ars1=(const float*)d_in[19];
  const float *Ws2=(const float*)d_in[20], *als2=(const float*)d_in[21], *ars2=(const float*)d_in[22];
  const float *Wres_g=(const float*)d_in[23], *Wres_s=(const float*)d_in[24];
  const float *Wout=(const float*)d_in[25], *bout=(const float*)d_in[26];

  // ---- carve workspace (~220 MB) ----
  char* p = (char*)d_ws;
  auto take = [&](size_t bytes) { void* r = (void*)p; p += alignup(bytes); return r; };
  Scratch ws;
  ws.f      = (float*)take((size_t)NNODES * 1024 * 4);
  ws.res    = (float*)take((size_t)NNODES * 1024 * 4);
  ws.bufA   = (float*)take((size_t)NNODES * 256 * 4);
  ws.bufB   = (float*)take((size_t)NNODES * 256 * 4);
  ws.el     = (float*)take((size_t)NNODES * NHEADS * 4);
  ws.er     = (float*)take((size_t)NNODES * NHEADS * 4);
  ws.den    = (float*)take((size_t)NNODES * NHEADS * 4);
  ws.ebuf   = (float*)take((size_t)NEDGES * NHEADS * 4);
  ws.abuf   = (float*)take((size_t)NEDGES * NHEADS * 4);
  ws.cnt    = (int*)take((size_t)NNODES * 4);
  ws.rs_g   = (int*)take((size_t)(NNODES + 1) * 4);
  ws.eidx_g = (int*)take((size_t)NEDGES * 4);
  ws.rs_s   = (int*)take((size_t)(NNODES + 1) * 4);
  ws.eidx_s = (int*)take((size_t)NEDGES * 4);
  ws.wt     = (__bf16*)take((size_t)1024 * 1024 * 2);

  // ---- CSR-by-dst for both edge sets (reused by all 3 layers) ----
  build_csr(dst_g, ws.cnt, ws.rs_g, ws.eidx_g, stream);
  build_csr(dst_s, ws.cnt, ws.rs_s, ws.eidx_s, stream);

  // ---- layer 0: add(elu(gat_g), elu(gat_s)), D=64, K=256 ----
  run_branch(x, 256, 64, Wg0, alg0, arg0, nullptr, src_g, dst_g, ws.rs_g, ws.eidx_g,
             ws.bufA, 1.0f, 0, 1, ws, stream);
  run_branch(x, 256, 64, Ws0, als0, ars0, nullptr, src_s, dst_s, ws.rs_s, ws.eidx_s,
             ws.bufA, 1.0f, 1, 1, ws, stream);
  // ---- layer 1: 0.5*(elu(gat_g)+elu(gat_s)), D=64, K=256 ----
  run_branch(ws.bufA, 256, 64, Wg1, alg1, arg1, nullptr, src_g, dst_g, ws.rs_g, ws.eidx_g,
             ws.bufB, 0.5f, 0, 1, ws, stream);
  run_branch(ws.bufA, 256, 64, Ws1, als1, ars1, nullptr, src_s, dst_s, ws.rs_s, ws.eidx_s,
             ws.bufB, 0.5f, 1, 1, ws, stream);
  // ---- layer 2: residual, no activation, D=256, K=256 ; h -> d_out tail ----
  float* hout = (float*)d_out + (size_t)NNODES * 40;
  run_branch(ws.bufB, 256, 256, Wg2, alg2, arg2, Wres_g, src_g, dst_g, ws.rs_g, ws.eidx_g,
             hout, 1.0f, 0, 0, ws, stream);
  run_branch(ws.bufB, 256, 256, Ws2, als2, ars2, Wres_s, src_s, dst_s, ws.rs_s, ws.eidx_s,
             hout, 1.0f, 1, 0, ws, stream);
  // ---- output head: h_final = h @ Wout + bout  (40 cols, padded tile) ----
  wtrans_kernel<<<g1((long)64 * 1024), 256, 0, stream>>>(Wout, ws.wt, 1024, 40, 64);
  gemm_bf16_wmma<<<dim3(NNODES / 32, 1), 256, 0, stream>>>(hout, ws.wt, (float*)d_out, bout,
                                                           1024, 40, 40);
}
